// VideoMambaTeacher_83897891160363
// MI455X (gfx1250) — compile-verified, matrix operand order checked by reading, not run on hardware
//
#include <hip/hip_runtime.h>
#include <hip/hip_bf16.h>

// ---------------------------------------------------------------------------
// VideoMamba forward for MI455X (gfx1250, wave32, WMMA).
// Big GEMMs: f16 inputs, f32 accumulate via v_wmma_f32_16x16x32_f16.
// Scan / conv / norms: f32 VALU.
// ---------------------------------------------------------------------------

typedef __attribute__((ext_vector_type(16))) _Float16 v16h;
typedef __attribute__((ext_vector_type(8)))  float    v8f;
typedef __attribute__((ext_vector_type(4)))  unsigned u32x4;

#define MP    1600            // padded rows (B*L = 1576 -> 1600)
#define TOK   1576            // B * L
#define LSEQ  197
#define DM    576
#define DIN   1152
#define XZC   2304            // 2 * DIN
#define DBLC  128             // padded (36 + 16 + 16 = 68 -> 128)
#define DTR   36
#define NST   16

union FragU { v16h h; u32x4 q[2]; };

__device__ __forceinline__ v16h load_frag(const _Float16* __restrict__ base, int k) {
  FragU f;
  f.q[0] = *reinterpret_cast<const u32x4*>(base + k);
  f.q[1] = *reinterpret_cast<const u32x4*>(base + k + 16);
  return f.h;
}

// C[M,N] = A[M,K] * W[N,K]^T   (A, W row-major f16; C row-major f32)
// Wave computes 32(M) x 64(N); block = 8 waves stacked in M.
__global__ __launch_bounds__(256) void gemm_f16_kernel(
    const _Float16* __restrict__ A, const _Float16* __restrict__ W,
    float* __restrict__ C, int M, int N, int K) {
  const int lane = threadIdx.x & 31;
  const int wave = threadIdx.x >> 5;
  const int r    = lane & 15;
  const int hi   = lane >> 4;
  const int mBase = (blockIdx.y * 8 + wave) * 32;
  if (mBase >= M) return;                       // wave-uniform: EXEC stays full
  const int nBase = blockIdx.x * 64;

  const _Float16* a0 = A + (size_t)(mBase + r) * K + hi * 8;
  const _Float16* a1 = a0 + (size_t)16 * K;
  const _Float16* w0 = W + (size_t)(nBase + r) * K + hi * 8;

  v8f acc[2][4] = {};
  for (int k = 0; k < K; k += 32) {
    __builtin_prefetch(a0 + k + 32, 0, 1);
    __builtin_prefetch(a1 + k + 32, 0, 1);
    v16h af0 = load_frag(a0, k);
    v16h af1 = load_frag(a1, k);
#pragma unroll
    for (int j = 0; j < 4; ++j) {
      v16h bf = load_frag(w0 + (size_t)(j * 16) * K, k);
      acc[0][j] = __builtin_amdgcn_wmma_f32_16x16x32_f16(
          false, af0, false, bf, (short)0, acc[0][j], false, false);
      acc[1][j] = __builtin_amdgcn_wmma_f32_16x16x32_f16(
          false, af1, false, bf, (short)0, acc[1][j], false, false);
    }
  }
  // C/D layout: VGPR v -> (M = v + 8*hi, N = lane&15)
#pragma unroll
  for (int i = 0; i < 2; ++i) {
#pragma unroll
    for (int j = 0; j < 4; ++j) {
      float* cp = C + (size_t)(mBase + i * 16 + hi * 8) * N + nBase + j * 16 + r;
#pragma unroll
      for (int v = 0; v < 8; ++v) cp[(size_t)v * N] = acc[i][j][v];
    }
  }
}

// ---------------------------------------------------------------------------

__global__ void cvt_f16_kernel(const float* __restrict__ in,
                               _Float16* __restrict__ out, size_t n) {
  size_t i = (size_t)blockIdx.x * blockDim.x + threadIdx.x;
  if (i < n) out[i] = (_Float16)in[i];
}

// x_proj weight slice (68 x DIN) -> zero-padded (128 x DIN) f16
__global__ void cvt_pad_xproj_kernel(const float* __restrict__ in,
                                     _Float16* __restrict__ out) {
  size_t i = (size_t)blockIdx.x * blockDim.x + threadIdx.x;
  if (i >= (size_t)DBLC * DIN) return;
  int row = (int)(i / DIN);
  out[i] = (row < 68) ? (_Float16)in[i] : (_Float16)0.0f;
}

// im2col for 16x16/str16 patch conv: rows = token index (b*197 + l), cols = 768
__global__ void im2col_kernel(const float* __restrict__ x, _Float16* __restrict__ out) {
  size_t idx = (size_t)blockIdx.x * blockDim.x + threadIdx.x;
  if (idx >= (size_t)MP * 768) return;
  int t = (int)(idx / 768), c = (int)(idx % 768);
  float v = 0.0f;
  if (t < TOK) {
    int b = t / LSEQ, l = t % LSEQ;
    if (l > 0) {
      int p = l - 1, py = p / 14, px = p % 14;
      int cin = c / 256, rem = c % 256, ky = rem / 16, kx = rem % 16;
      v = x[(((size_t)b * 3 + cin) * 224 + (py * 16 + ky)) * 224 + (px * 16 + kx)];
    }
  }
  out[idx] = (_Float16)v;
}

__global__ void assemble_h_kernel(const float* __restrict__ tok,
                                  const float* __restrict__ patch_b,
                                  const float* __restrict__ cls,
                                  const float* __restrict__ pos,
                                  float* __restrict__ h, float* __restrict__ res) {
  size_t idx = (size_t)blockIdx.x * blockDim.x + threadIdx.x;
  if (idx >= (size_t)TOK * DM) return;
  int t = (int)(idx / DM), d = (int)(idx % DM), l = t % LSEQ;
  float v = (l == 0) ? cls[d] : (tok[idx] + patch_b[d]);
  h[idx]   = v + pos[(size_t)l * DM + d];
  res[idx] = 0.0f;
}

// rr = hh + rr; out = rmsnorm(rr)*w ; writes f16 and/or f32 outputs.
__global__ __launch_bounds__(256) void add_rmsnorm_kernel(
    const float* __restrict__ hh, float* __restrict__ rr,
    const float* __restrict__ w, _Float16* __restrict__ out16,
    float* __restrict__ out32) {
  const size_t base = (size_t)blockIdx.x * DM;
  __shared__ float sm[256];
  float s = 0.0f;
  for (int c = threadIdx.x; c < DM; c += 256) {
    float v = hh[base + c] + rr[base + c];
    rr[base + c] = v;
    s += v * v;
  }
  sm[threadIdx.x] = s;
  __syncthreads();
  for (int off = 128; off > 0; off >>= 1) {
    if ((int)threadIdx.x < off) sm[threadIdx.x] += sm[threadIdx.x + off];
    __syncthreads();
  }
  const float scale = rsqrtf(sm[0] / (float)DM + 1e-5f);
  for (int c = threadIdx.x; c < DM; c += 256) {
    float v = rr[base + c] * scale * w[c];
    if (out16) out16[base + c] = (_Float16)v;
    if (out32) out32[base + c] = v;
  }
}

// causal depthwise conv (K=4, left pad 3) + SiLU, with optional seq flip.
// input: x-part of xz (cols 0..DIN-1). one thread per (b, d).
__global__ void conv_silu_kernel(const float* __restrict__ xz,
                                 const float* __restrict__ cw,
                                 const float* __restrict__ cb,
                                 float* __restrict__ xout,
                                 _Float16* __restrict__ xout16, int flip) {
  int idx = blockIdx.x * blockDim.x + threadIdx.x;
  if (idx >= 8 * DIN) return;
  int b = idx / DIN, d = idx % DIN;
  float w0 = cw[d * 4 + 0], w1 = cw[d * 4 + 1], w2 = cw[d * 4 + 2], w3 = cw[d * 4 + 3];
  float bias = cb[d];
  float x0 = 0.f, x1 = 0.f, x2 = 0.f;
  for (int s = 0; s < LSEQ; ++s) {
    int l = flip ? (LSEQ - 1 - s) : s;
    float xs = xz[(size_t)(b * LSEQ + l) * XZC + d];
    float y = w0 * x0 + w1 * x1 + w2 * x2 + w3 * xs + bias;
    x0 = x1; x1 = x2; x2 = xs;
    y = y / (1.0f + expf(-y));                 // SiLU
    size_t o = (size_t)(b * LSEQ + s) * DIN + d;
    xout[o]   = y;
    xout16[o] = (_Float16)y;
  }
}

// delta = softplus(dtp_w @ dt + dtp_b), dt = dbl[:, 0:36]
__global__ void delta_softplus_kernel(const float* __restrict__ dbl,
                                      const float* __restrict__ dpw,
                                      const float* __restrict__ dpb,
                                      float* __restrict__ delta) {
  size_t idx = (size_t)blockIdx.x * blockDim.x + threadIdx.x;
  if (idx >= (size_t)TOK * DIN) return;
  int t = (int)(idx / DIN), d = (int)(idx % DIN);
  const float* dt = dbl + (size_t)t * DBLC;
  const float* wr = dpw + (size_t)d * DTR;
  float a = dpb[d];
#pragma unroll 4
  for (int r = 0; r < DTR; ++r) a += wr[r] * dt[r];
  delta[idx] = (a > 20.0f) ? a : log1pf(expf(a));
}

// selective scan: one thread per (b, d), 16-state recurrence over L.
// inputs are in branch order; output scattered back to original order.
__global__ void scan_kernel(const float* __restrict__ u,
                            const float* __restrict__ delta,
                            const float* __restrict__ dbl,
                            const float* __restrict__ xz,   // for z gate
                            const float* __restrict__ a_log,
                            const float* __restrict__ dvec,
                            float* __restrict__ ysum, int flip, int first) {
  int idx = blockIdx.x * blockDim.x + threadIdx.x;
  if (idx >= 8 * DIN) return;
  int b = idx / DIN, d = idx % DIN;
  float A[NST], h[NST];
#pragma unroll
  for (int n = 0; n < NST; ++n) { A[n] = -expf(a_log[(size_t)d * NST + n]); h[n] = 0.f; }
  const float Dd = dvec[d];
  for (int s = 0; s < LSEQ; ++s) {
    size_t tb = (size_t)(b * LSEQ + s);
    float dlt = delta[tb * DIN + d];
    float uu  = u[tb * DIN + d];
    const float* BC = dbl + tb * DBLC;
    float du = dlt * uu;
    float y = 0.f;
#pragma unroll
    for (int n = 0; n < NST; ++n) {
      h[n] = expf(dlt * A[n]) * h[n] + du * BC[DTR + n];
      y += h[n] * BC[DTR + NST + n];
    }
    int l = flip ? (LSEQ - 1 - s) : s;
    float zv = xz[(size_t)(b * LSEQ + l) * XZC + DIN + d];
    y = (y + Dd * uu) * (zv / (1.0f + expf(-zv)));
    size_t o = (size_t)(b * LSEQ + l) * DIN + d;
    if (first) ysum[o] = y; else ysum[o] += y;
  }
}

__global__ void head_kernel(const float* __restrict__ obuf,
                            const float* __restrict__ hw,
                            const float* __restrict__ hb,
                            float* __restrict__ out) {
  int idx = blockIdx.x * blockDim.x + threadIdx.x;
  if (idx >= 8 * 1000) return;
  int b = idx / 1000, c = idx % 1000;
  const float* row = obuf + (size_t)(b * LSEQ) * DM;
  const float* w   = hw + (size_t)c * DM;
  float a = hb[c];
  for (int d = 0; d < DM; ++d) a += row[d] * w[d];
  out[idx] = a;
}

__global__ void write_patch_kernel(const float* __restrict__ obuf,
                                   float* __restrict__ dst) {
  size_t idx = (size_t)blockIdx.x * blockDim.x + threadIdx.x;
  if (idx >= (size_t)8 * 196 * DM) return;
  int d = (int)(idx % DM);
  size_t bp = idx / DM;
  int b = (int)(bp / 196), p = (int)(bp % 196);
  dst[idx] = obuf[(size_t)(b * LSEQ + p + 1) * DM + d];
}

// ---------------------------------------------------------------------------

extern "C" void kernel_launch(void* const* d_in, const int* in_sizes, int n_in,
                              void* d_out, int out_size, void* d_ws, size_t ws_size,
                              hipStream_t stream) {
  (void)in_sizes; (void)n_in; (void)out_size; (void)ws_size;
  const float* x          = (const float*)d_in[0];
  const float* patch_w    = (const float*)d_in[1];
  const float* patch_b    = (const float*)d_in[2];
  const float* cls_token  = (const float*)d_in[3];
  const float* pos_embed  = (const float*)d_in[4];
  const float* norm_ws    = (const float*)d_in[5];
  const float* in_proj_w  = (const float*)d_in[6];
  const float* out_proj_w = (const float*)d_in[7];
  const float* conv_ws    = (const float*)d_in[8];
  const float* conv_bs    = (const float*)d_in[9];
  const float* xproj_ws   = (const float*)d_in[10];
  const float* dtproj_ws  = (const float*)d_in[11];
  const float* dtproj_bs  = (const float*)d_in[12];
  const float* A_logs     = (const float*)d_in[13];
  const float* Ds         = (const float*)d_in[14];
  const float* conv_b_ws  = (const float*)d_in[15];
  const float* conv_b_bs  = (const float*)d_in[16];
  const float* xproj_b_ws = (const float*)d_in[17];
  const float* dtproj_b_w = (const float*)d_in[18];
  const float* dtproj_b_b = (const float*)d_in[19];
  const float* A_b_logs   = (const float*)d_in[20];
  const float* D_bs       = (const float*)d_in[21];
  const float* normf_w    = (const float*)d_in[22];
  const float* head_w     = (const float*)d_in[23];
  const float* head_b     = (const float*)d_in[24];

  char* ws = (char*)d_ws;
  size_t off = 0;
  auto alloc = [&](size_t bytes) -> void* {
    void* p = ws + off;
    off = (off + bytes + 255) & ~(size_t)255;
    return p;
  };
  float*    h_buf  = (float*)alloc((size_t)MP * DM * 4);
  float*    res    = (float*)alloc((size_t)MP * DM * 4);
  float*    xz     = (float*)alloc((size_t)MP * XZC * 4);
  float*    xbuf   = (float*)alloc((size_t)MP * DIN * 4);
  float*    delta  = (float*)alloc((size_t)MP * DIN * 4);
  float*    dbl    = (float*)alloc((size_t)MP * DBLC * 4);
  float*    ysum   = (float*)alloc((size_t)MP * DIN * 4);
  float*    obuf   = (float*)alloc((size_t)MP * DM * 4);
  _Float16* xim    = (_Float16*)alloc((size_t)MP * 768 * 2);
  _Float16* xn16   = (_Float16*)alloc((size_t)MP * DM * 2);
  _Float16* xc16   = (_Float16*)alloc((size_t)MP * DIN * 2);
  _Float16* y16    = (_Float16*)alloc((size_t)MP * DIN * 2);
  _Float16* w_in16 = (_Float16*)alloc((size_t)XZC * DM * 2);
  _Float16* w_x16  = (_Float16*)alloc((size_t)DBLC * DIN * 2);
  _Float16* w_o16  = (_Float16*)alloc((size_t)DM * DIN * 2);
  _Float16* w_p16  = (_Float16*)alloc((size_t)DM * 768 * 2);

  const dim3 blk(256);
  auto b1d = [](size_t n) { return dim3((unsigned)((n + 255) / 256)); };
  const unsigned gy = (MP + 255) / 256;   // GEMM grid.y (8 waves * 32 rows)

  // ---- patch embed ----
  cvt_f16_kernel<<<b1d((size_t)DM * 768), blk, 0, stream>>>(patch_w, w_p16, (size_t)DM * 768);
  im2col_kernel<<<b1d((size_t)MP * 768), blk, 0, stream>>>(x, xim);
  gemm_f16_kernel<<<dim3(DM / 64, gy), blk, 0, stream>>>(xim, w_p16, obuf, MP, DM, 768);
  assemble_h_kernel<<<b1d((size_t)TOK * DM), blk, 0, stream>>>(obuf, patch_b, cls_token,
                                                               pos_embed, h_buf, res);

  // ---- 24 bimamba layers ----
  for (int layer = 0; layer < 24; ++layer) {
    const float* nw  = norm_ws + (size_t)layer * DM;
    const float* ipw = in_proj_w + (size_t)layer * XZC * DM;
    const float* opw = out_proj_w + (size_t)layer * DM * DIN;

    add_rmsnorm_kernel<<<TOK, blk, 0, stream>>>(h_buf, res, nw, xn16, nullptr);
    cvt_f16_kernel<<<b1d((size_t)XZC * DM), blk, 0, stream>>>(ipw, w_in16, (size_t)XZC * DM);
    gemm_f16_kernel<<<dim3(XZC / 64, gy), blk, 0, stream>>>(xn16, w_in16, xz, MP, XZC, DM);

    for (int br = 0; br < 2; ++br) {
      const float* cw  = (br ? conv_b_ws : conv_ws) + (size_t)layer * DIN * 4;
      const float* cb  = (br ? conv_b_bs : conv_bs) + (size_t)layer * DIN;
      const float* xpw = (br ? xproj_b_ws : xproj_ws) + (size_t)layer * 68 * DIN;
      const float* dpw = (br ? dtproj_b_w : dtproj_ws) + (size_t)layer * DIN * DTR;
      const float* dpb = (br ? dtproj_b_b : dtproj_bs) + (size_t)layer * DIN;
      const float* al  = (br ? A_b_logs : A_logs) + (size_t)layer * DIN * NST;
      const float* dv  = (br ? D_bs : Ds) + (size_t)layer * DIN;

      cvt_pad_xproj_kernel<<<b1d((size_t)DBLC * DIN), blk, 0, stream>>>(xpw, w_x16);
      conv_silu_kernel<<<36, blk, 0, stream>>>(xz, cw, cb, xbuf, xc16, br);
      gemm_f16_kernel<<<dim3(DBLC / 64, gy), blk, 0, stream>>>(xc16, w_x16, dbl, MP, DBLC, DIN);
      delta_softplus_kernel<<<b1d((size_t)TOK * DIN), blk, 0, stream>>>(dbl, dpw, dpb, delta);
      scan_kernel<<<36, blk, 0, stream>>>(xbuf, delta, dbl, xz, al, dv, ysum, br, br == 0);
    }

    cvt_f16_kernel<<<b1d((size_t)TOK * DIN), blk, 0, stream>>>(ysum, y16, (size_t)TOK * DIN);
    cvt_f16_kernel<<<b1d((size_t)DM * DIN), blk, 0, stream>>>(opw, w_o16, (size_t)DM * DIN);
    gemm_f16_kernel<<<dim3(DM / 64, gy), blk, 0, stream>>>(y16, w_o16, h_buf, MP, DM, DIN);
  }

  // ---- final add-norm, head, patch outputs ----
  add_rmsnorm_kernel<<<TOK, blk, 0, stream>>>(h_buf, res, normf_w, nullptr, obuf);
  head_kernel<<<b1d(8 * 1000), blk, 0, stream>>>(obuf, head_w, head_b, (float*)d_out);
  write_patch_kernel<<<b1d((size_t)8 * 196 * DM), blk, 0, stream>>>(obuf, (float*)d_out + 8000);
}
